// BigramLanguageModel_50732153701075
// MI455X (gfx1250) — matrix operand order repaired and checked by reading, not compiled, not run on hardware
//
#include <hip/hip_runtime.h>
#include <math.h>

typedef __attribute__((ext_vector_type(16))) _Float16 v16h;
typedef __attribute__((ext_vector_type(8)))  float    v8f;
typedef __attribute__((ext_vector_type(4)))  unsigned v4u;
typedef __attribute__((ext_vector_type(8)))  int      v8i_;
typedef __attribute__((ext_vector_type(4)))  int      v4i_;

namespace cfg {
constexpr int V = 65, B = 128, T = 256, C = 384, H = 6, L = 6;
constexpr int HS = C / H;    // 64
constexpr int FF = 4 * C;    // 1536
constexpr int BT = B * T;    // 32768
}

union AFrag { v16h v; unsigned u[8]; _Float16 h[16]; };
union BFrag { v16h v; unsigned u[8]; _Float16 h[16]; };

__device__ __forceinline__ v8f wmma_f16(const AFrag& a, const BFrag& b, v8f c) {
  return __builtin_amdgcn_wmma_f32_16x16x32_f16(false, a.v, false, b.v, (short)0, c, false, false);
}

// ---------------- weight conversion (f32 -> f16, stored TRANSPOSED [N][K]) ----------------
// generic: in [K][N] f32 -> out [N][K] f16
__global__ void conv_t_kernel(const float* __restrict__ in, _Float16* __restrict__ out,
                              int K, int N) {
  int i = blockIdx.x * 256 + threadIdx.x;
  if (i >= K * N) return;
  int n = i / K, k = i % K;
  out[i] = (_Float16)in[(size_t)k * N + n];
}

// wq/wk/wv: in (L,H,C,HS) f32 -> out (L, N=C, K=C) f16, out[l][h*HS+d][c] = in[l][h][c][d]
__global__ void conv_qkv_t_kernel(const float* __restrict__ in, _Float16* __restrict__ out) {
  using namespace cfg;
  int i = blockIdx.x * 256 + threadIdx.x;
  if (i >= L * C * C) return;
  int l = i / (C * C);
  int rem = i % (C * C);
  int n = rem / C;     // h*HS + d
  int c = rem % C;     // K index
  int h = n / HS, d = n % HS;
  out[i] = (_Float16)in[(((size_t)l * H + h) * C + c) * HS + d];
}

// ---------------- embedding ----------------
__global__ void embed_kernel(const float* __restrict__ tok, const float* __restrict__ pos,
                             const int* __restrict__ idx, float* __restrict__ x) {
  using namespace cfg;
  size_t i = (size_t)blockIdx.x * 256 + threadIdx.x;
  if (i >= (size_t)BT * C) return;
  int c = (int)(i % C);
  size_t bt = i / C;
  int t = (int)(bt % T);
  int tk = idx[bt];
  x[i] = tok[(size_t)tk * C + c] + pos[(size_t)t * C + c];
}

// ---------------- layernorm (f32 in -> f16 out), one wave32 per row of 384 ----------------
__global__ void ln_kernel(const float* __restrict__ x, const float* __restrict__ s,
                          const float* __restrict__ bb, _Float16* __restrict__ out) {
  using namespace cfg;
  const int wave = threadIdx.x >> 5, lane = threadIdx.x & 31;
  const size_t row = (size_t)blockIdx.x * 4 + wave;
  const float* xr = x + row * C;
  float v[12];
  float sum = 0.f, sq = 0.f;
#pragma unroll
  for (int i = 0; i < 12; ++i) {
    float t = xr[lane + 32 * i];
    v[i] = t; sum += t; sq += t * t;
  }
#pragma unroll
  for (int off = 16; off > 0; off >>= 1) {
    sum += __shfl_xor(sum, off, 32);
    sq  += __shfl_xor(sq,  off, 32);
  }
  const float mean = sum * (1.f / C);
  const float var = sq * (1.f / C) - mean * mean;
  const float rs = rsqrtf(var + 1e-5f);
#pragma unroll
  for (int i = 0; i < 12; ++i) {
    int c = lane + 32 * i;
    out[row * C + c] = (_Float16)((v[i] - mean) * rs * s[c] + bb[c]);
  }
}

// ---------------- generic WMMA GEMM: out = A[M,K](f16) x B (f16, stored [N][K]) (+bias) ----
// MODE 0: f16 out; MODE 1: f16 relu out; MODE 2: f32 residual in-place;
// MODE 3: f32 out with column guard (head, N=65);
// MODE 4: f16 out transposed per-(b,h) for attention V: vt[((b*H+h)*HS+d)*T + t]
constexpr int KC = 128;        // K-chunk staged in LDS per TDM load
constexpr int LROW = KC + 2;   // LDS row stride in halves (TDM pad: 1 dword per 64-dword row)

template <int MODE>
__global__ void gemm_kernel(const _Float16* __restrict__ A, const _Float16* __restrict__ Bt,
                            const float* __restrict__ bias,
                            float* __restrict__ outf, _Float16* __restrict__ outh,
                            int N, int K) {
  __shared__ _Float16 sB[64 * LROW];
  const int tid = threadIdx.x;
  const int wave = tid >> 5, lane = tid & 31;
  const int n0 = blockIdx.x * 64;
  const size_t row0 = (size_t)blockIdx.y * 64 + wave * 16;
  const int m = lane & 15, hf = lane >> 4;
  const int kbA = hf * 16, kbB = hf * 16;
  const _Float16* Arow = A + (row0 + m) * (size_t)K;

  const v8f vz = {0.f, 0.f, 0.f, 0.f, 0.f, 0.f, 0.f, 0.f};
  v8f acc[4];
#pragma unroll
  for (int nt = 0; nt < 4; ++nt) acc[nt] = vz;

  for (int kc = 0; kc < K; kc += KC) {
#if __has_builtin(__builtin_amdgcn_tensor_load_to_lds)
    // Tensor Data Mover: stage B panel [64 rows x KC halves] -> LDS, padded rows.
    if (wave == 0) {
      unsigned lds = (unsigned)(unsigned long long)(const void*)&sB[0];
      unsigned long long ga =
          (unsigned long long)(const void*)(Bt + (size_t)n0 * K + kc);
      int rows = N - n0;               // OOB rows (head matrix) read back zero
      if (rows > 64 * 1024 - 1) rows = 64 * 1024 - 1;
      v4u g0 = {1u,                    // count=1 (valid), user mode
                lds,                   // lds_addr
                (unsigned)ga,          // global_addr lo
                (unsigned)(ga >> 32) | (2u << 30)};  // global_addr hi | type=2
      v8i_ g1 = {(int)((1u << 16) | (1u << 20) | (5u << 22)),  // data_size=2B, pad_en, pad_interval=64dw (pad 1dw)
                 (int)(KC << 16),                    // tensor_dim0 lo16 (atomic_barrier_addr=0)
                 (int)((unsigned)rows << 16),        // tensor_dim0 hi16=0 | tensor_dim1 lo16
                 (int)(KC << 16),                    // tensor_dim1 hi16=0 | tile_dim0=KC
                 64,                                 // tile_dim1=64, tile_dim2=0
                 K,                                  // tensor_dim0_stride lo32 (elements)
                 0, 0};                              // strides hi / tensor_dim1_stride
      v4i_ gz = {0, 0, 0, 0};
#if __has_include(<hip/amd_detail/amd_gfx1250_TDM.h>)
      v8i_ g4 = {0, 0, 0, 0, 0, 0, 0, 0};
      __builtin_amdgcn_tensor_load_to_lds(g0, g1, gz, gz, g4, 0);
#else
      __builtin_amdgcn_tensor_load_to_lds(g0, g1, gz, gz, 0);
#endif
      __builtin_amdgcn_s_wait_tensorcnt(0);
    }
#else
    // fallback: cooperative dword loads (B is [N][K] so K-pairs are contiguous)
    for (int i = 0; i < 32; ++i) {
      int di = tid * 32 + i;            // dword index over 64*KC/2
      int n = di >> 6;                  // row (KC/2 = 64 dwords per row)
      int kk = (di & 63) << 1;          // half offset within row
      unsigned val = 0;
      if (n0 + n < N) val = *(const unsigned*)(Bt + (size_t)(n0 + n) * K + kc + kk);
      *(unsigned*)(&sB[n * LROW + kk]) = val;
    }
#endif
    __syncthreads();

    if (kc + KC < K)  // gfx1250 global prefetch of next A chunk
      __builtin_prefetch(Arow + kc + KC + (tid & 3) * 32, 0, 1);

    for (int k0 = 0; k0 < KC; k0 += 32) {
      AFrag a;
#pragma unroll
      for (int j = 0; j < 4; ++j) {
        a.u[j]     = *(const unsigned*)(Arow + kc + k0 + kbA + 2 * j);
        a.u[j + 4] = *(const unsigned*)(Arow + kc + k0 + 8 + kbA + 2 * j);
      }
#pragma unroll
      for (int nt = 0; nt < 4; ++nt) {
        BFrag bf;
        const _Float16* p = &sB[(nt * 16 + m) * LROW + k0 + kbB];
#pragma unroll
        for (int j = 0; j < 8; ++j) bf.u[j] = *(const unsigned*)(p + 2 * j);
        acc[nt] = wmma_f16(a, bf, acc[nt]);
      }
    }
    __syncthreads();
  }

#pragma unroll
  for (int nt = 0; nt < 4; ++nt) {
#pragma unroll
    for (int j = 0; j < 8; ++j) {
      const size_t r = row0 + j + 8 * hf;
      const int n = n0 + nt * 16 + m;
      float val = acc[nt][j];
      if (MODE == 3) {
        if (n < N) {
          if (bias) val += bias[n];
          outf[r * (size_t)N + n] = val;
        }
      } else if (MODE == 4) {
        // transposed V: vt[((b*H + h)*HS + d)*T + t]
        int b = (int)(r / cfg::T), t = (int)(r % cfg::T);
        int h = n / cfg::HS, d = n % cfg::HS;
        outh[(((size_t)b * cfg::H + h) * cfg::HS + d) * cfg::T + t] = (_Float16)val;
      } else {
        if (bias) val += bias[n];
        if (MODE == 0) outh[r * (size_t)N + n] = (_Float16)val;
        else if (MODE == 1) outh[r * (size_t)N + n] = (_Float16)fmaxf(val, 0.f);
        else outf[r * (size_t)N + n] = outf[r * (size_t)N + n] + val;  // residual
      }
    }
  }
}

// ---------------- fused causal attention: one block = (b, h, 64 q-rows) ----------------
// q,k: f16 [BT, C] (column block h*HS); vt: f16 [B,H,HS,T] (transposed)
__global__ void attn_kernel(const _Float16* __restrict__ q, const _Float16* __restrict__ k,
                            const _Float16* __restrict__ vt, _Float16* __restrict__ att) {
  using namespace cfg;
  extern __shared__ char smem[];
  float* S = (float*)smem;                          // [64][256] scores
  _Float16* P = (_Float16*)(smem + 64 * 256 * 4);   // [64][256] probs
  const int tid = threadIdx.x;
  const int wave = tid >> 5, lane = tid & 31;
  const int qblk = blockIdx.x & 3;
  const int bh = blockIdx.x >> 2;
  const int b = bh / H, h = bh % H;
  const int q0 = qblk * 64;
  const _Float16* qptr = q + ((size_t)(b * T + q0)) * C + h * HS;
  const _Float16* kptr = k + ((size_t)b * T) * C + h * HS;
  const _Float16* vtb = vt + ((size_t)(b * H + h) * HS) * T;
  _Float16* optr = att + ((size_t)(b * T + q0)) * C + h * HS;

  const int m = lane & 15, hf = lane >> 4;
  const int kbA = hf * 16, kbB = hf * 16;
  const float scale = 0.05103103630798288f;  // C^-0.5 (reference scales by n_embed)

  // this wave's Q fragments (rows wave*16..+15, d split 0..31 / 32..63)
  AFrag aq0, aq1;
  {
    const _Float16* p = qptr + (size_t)(wave * 16 + m) * C;
#pragma unroll
    for (int j = 0; j < 4; ++j) {
      aq0.u[j]     = *(const unsigned*)(p + kbA + 2 * j);
      aq0.u[j + 4] = *(const unsigned*)(p + 8 + kbA + 2 * j);
      aq1.u[j]     = *(const unsigned*)(p + 32 + kbA + 2 * j);
      aq1.u[j + 4] = *(const unsigned*)(p + 40 + kbA + 2 * j);
    }
  }

  const int stmax = (q0 >> 4) + wave;  // last s-tile in causal region (wave-uniform)
  for (int st = 0; st < 16; ++st) {
    if (st <= stmax) {
      BFrag b0, b1;
      const int s = st * 16 + m;  // lane column = key index s; K dim = d (contiguous)
      const _Float16* p = kptr + (size_t)s * C + kbB;
#pragma unroll
      for (int j = 0; j < 8; ++j) {
        b0.u[j] = *(const unsigned*)(p + 2 * j);
        b1.u[j] = *(const unsigned*)(p + 32 + 2 * j);
      }
      v8f acc = {0.f, 0.f, 0.f, 0.f, 0.f, 0.f, 0.f, 0.f};
      acc = wmma_f16(aq0, b0, acc);
      acc = wmma_f16(aq1, b1, acc);
#pragma unroll
      for (int j = 0; j < 8; ++j) {
        int tl = wave * 16 + j + 8 * hf;
        int sc = st * 16 + m;
        S[tl * 256 + sc] = (sc <= q0 + tl) ? acc[j] * scale : -INFINITY;
      }
    } else {
#pragma unroll
      for (int j = 0; j < 8; ++j)
        S[(wave * 16 + j + 8 * hf) * 256 + st * 16 + m] = -INFINITY;
    }
  }
  __syncthreads();

  // row-wise softmax (64 rows), f16 probs into P
  if (tid < 64) {
    const int r = tid;
    float mx = -INFINITY;
    for (int s = 0; s < 256; ++s) mx = fmaxf(mx, S[r * 256 + s]);
    float sum = 0.f;
    for (int s = 0; s < 256; ++s) sum += __expf(S[r * 256 + s] - mx);
    float inv = 1.f / sum;
    for (int s = 0; s < 256; ++s) P[r * 256 + s] = (_Float16)(__expf(S[r * 256 + s] - mx) * inv);
  }
  __syncthreads();

  // att = P[64,256] x V[256,64] (V read transposed: contiguous K-pairs)
  const v8f vz = {0.f, 0.f, 0.f, 0.f, 0.f, 0.f, 0.f, 0.f};
  v8f o[4];
#pragma unroll
  for (int nt = 0; nt < 4; ++nt) o[nt] = vz;

  const int niter = (q0 >> 5) + 2;  // probs beyond q0+63 are zero
  for (int ki = 0; ki < niter; ++ki) {
    const int s0 = ki * 32;
    AFrag ap;
    {
      const _Float16* p = P + (size_t)(wave * 16 + m) * 256 + s0;
#pragma unroll
      for (int j = 0; j < 4; ++j) {
        ap.u[j]     = *(const unsigned*)(p + kbA + 2 * j);
        ap.u[j + 4] = *(const unsigned*)(p + 8 + kbA + 2 * j);
      }
    }
#pragma unroll
    for (int nt = 0; nt < 4; ++nt) {
      BFrag bv;
      const _Float16* p = vtb + (size_t)(nt * 16 + m) * T + s0 + kbB;
#pragma unroll
      for (int j = 0; j < 8; ++j) bv.u[j] = *(const unsigned*)(p + 2 * j);
      o[nt] = wmma_f16(ap, bv, o[nt]);
    }
  }
#pragma unroll
  for (int nt = 0; nt < 4; ++nt) {
#pragma unroll
    for (int j = 0; j < 8; ++j) {
      int tl = wave * 16 + j + 8 * hf;
      optr[(size_t)tl * C + nt * 16 + m] = (_Float16)o[nt][j];
    }
  }
}

// ---------------- loss ----------------
__global__ void init_loss_kernel(float* loss) {
  if (threadIdx.x == 0 && blockIdx.x == 0) loss[0] = 0.f;
}

__global__ void loss_kernel(const float* __restrict__ logits, const int* __restrict__ tgt,
                            float* __restrict__ loss) {
  using namespace cfg;
  int r = blockIdx.x * 256 + threadIdx.x;
  if (r >= BT) return;
  const float* row = logits + (size_t)r * V;
  float mx = -INFINITY;
  for (int j = 0; j < V; ++j) mx = fmaxf(mx, row[j]);
  float s = 0.f;
  for (int j = 0; j < V; ++j) s += __expf(row[j] - mx);
  float lse = logf(s) + mx;
  atomicAdd(loss, (lse - row[tgt[r]]) * (1.f / BT));
}

// ---------------- host launch ----------------
extern "C" void kernel_launch(void* const* d_in, const int* in_sizes, int n_in,
                              void* d_out, int out_size, void* d_ws, size_t ws_size,
                              hipStream_t stream) {
  using namespace cfg;
  (void)in_sizes; (void)n_in; (void)out_size; (void)ws_size;

  const float* tok_emb = (const float*)d_in[0];
  const float* pos_emb = (const float*)d_in[1];
  const float* wq     = (const float*)d_in[2];
  const float* wk     = (const float*)d_in[3];
  const float* wv     = (const float*)d_in[4];
  const float* proj_w = (const float*)d_in[5];
  const float* proj_b = (const float*)d_in[6];
  const float* ln1_s  = (const float*)d_in[7];
  const float* ln1_b  = (const float*)d_in[8];
  const float* w1     = (const float*)d_in[9];
  const float* b1     = (const float*)d_in[10];
  const float* w2     = (const float*)d_in[11];
  const float* b2     = (const float*)d_in[12];
  const float* ln2_s  = (const float*)d_in[13];
  const float* ln2_b  = (const float*)d_in[14];
  const float* lnf_s  = (const float*)d_in[15];
  const float* lnf_b  = (const float*)d_in[16];
  const float* head_w = (const float*)d_in[17];
  const float* head_b = (const float*)d_in[18];
  const int* idx      = (const int*)d_in[19];
  const int* targets  = (const int*)d_in[20];

  float* logits = (float*)d_out;
  float* loss = logits + (size_t)BT * V;

  char* ws = (char*)d_ws;
  size_t off = 0;
  auto alloc = [&](size_t bytes) -> char* {
    char* p = ws + off;
    off = (off + bytes + 255) & ~(size_t)255;
    return p;
  };
  float*     x    = (float*)    alloc((size_t)BT * C * 4);
  _Float16*  xn   = (_Float16*) alloc((size_t)BT * C * 2);
  _Float16*  qb   = (_Float16*) alloc((size_t)BT * C * 2);
  _Float16*  kb   = (_Float16*) alloc((size_t)BT * C * 2);
  _Float16*  vtb  = (_Float16*) alloc((size_t)BT * C * 2);   // transposed V [B,H,HS,T]
  _Float16*  attb = (_Float16*) alloc((size_t)BT * C * 2);
  _Float16*  hb   = (_Float16*) alloc((size_t)BT * FF * 2);
  _Float16*  wqt  = (_Float16*) alloc((size_t)L * C * C * 2);
  _Float16*  wkt  = (_Float16*) alloc((size_t)L * C * C * 2);
  _Float16*  wvt  = (_Float16*) alloc((size_t)L * C * C * 2);
  _Float16*  pwt  = (_Float16*) alloc((size_t)L * C * C * 2);
  _Float16*  w1t  = (_Float16*) alloc((size_t)L * C * FF * 2);
  _Float16*  w2t  = (_Float16*) alloc((size_t)L * FF * C * 2);
  _Float16*  hwt  = (_Float16*) alloc((size_t)V * C * 2);

  // weights -> f16, transposed to [N][K]
  {
    int n = L * C * C;
    dim3 g((n + 255) / 256);
    conv_qkv_t_kernel<<<g, 256, 0, stream>>>(wq, wqt);
    conv_qkv_t_kernel<<<g, 256, 0, stream>>>(wk, wkt);
    conv_qkv_t_kernel<<<g, 256, 0, stream>>>(wv, wvt);
  }
  for (int l = 0; l < L; ++l) {
    dim3 gc((C * C + 255) / 256), gf((C * FF + 255) / 256);
    conv_t_kernel<<<gc, 256, 0, stream>>>(proj_w + (size_t)l * C * C, pwt + (size_t)l * C * C, C, C);
    conv_t_kernel<<<gf, 256, 0, stream>>>(w1 + (size_t)l * C * FF, w1t + (size_t)l * C * FF, C, FF);
    conv_t_kernel<<<gf, 256, 0, stream>>>(w2 + (size_t)l * FF * C, w2t + (size_t)l * FF * C, FF, C);
  }
  conv_t_kernel<<<dim3((C * V + 255) / 256), 256, 0, stream>>>(head_w, hwt, C, V);

  embed_kernel<<<dim3(((size_t)BT * C + 255) / 256), 256, 0, stream>>>(tok_emb, pos_emb, idx, x);

  const dim3 blk(128);
  const size_t attn_lds = 64 * 256 * 4 + 64 * 256 * 2;  // 96 KB: scores f32 + probs f16
  for (int l = 0; l < L; ++l) {
    ln_kernel<<<BT / 4, 128, 0, stream>>>(x, ln1_s + l * C, ln1_b + l * C, xn);
    dim3 gC(C / 64, BT / 64);
    gemm_kernel<0><<<gC, blk, 0, stream>>>(xn, wqt + (size_t)l * C * C, nullptr, nullptr, qb, C, C);
    gemm_kernel<0><<<gC, blk, 0, stream>>>(xn, wkt + (size_t)l * C * C, nullptr, nullptr, kb, C, C);
    gemm_kernel<4><<<gC, blk, 0, stream>>>(xn, wvt + (size_t)l * C * C, nullptr, nullptr, vtb, C, C);
    attn_kernel<<<B * H * (T / 64), blk, attn_lds, stream>>>(qb, kb, vtb, attb);
    gemm_kernel<2><<<gC, blk, 0, stream>>>(attb, pwt + (size_t)l * C * C, proj_b + l * C, x, nullptr, C, C);
    ln_kernel<<<BT / 4, 128, 0, stream>>>(x, ln2_s + l * C, ln2_b + l * C, xn);
    gemm_kernel<1><<<dim3(FF / 64, BT / 64), blk, 0, stream>>>(xn, w1t + (size_t)l * C * FF, b1 + l * FF, nullptr, hb, FF, C);
    gemm_kernel<2><<<gC, blk, 0, stream>>>(hb, w2t + (size_t)l * FF * C, b2 + l * C, x, nullptr, C, FF);
  }
  ln_kernel<<<BT / 4, 128, 0, stream>>>(x, lnf_s, lnf_b, xn);
  gemm_kernel<3><<<dim3((V + 63) / 64, BT / 64), blk, 0, stream>>>(xn, hwt, head_b, logits, nullptr, V, C);
  init_loss_kernel<<<1, 1, 0, stream>>>(loss);
  loss_kernel<<<(BT + 255) / 256, 256, 0, stream>>>(logits, targets, loss);
}
